// MinkUNet_56470230008550
// MI455X (gfx1250) — matrix operand order repaired and checked by reading, not compiled, not run on hardware
//
#include <hip/hip_runtime.h>
#include <hip/hip_bf16.h>
#include <math.h>

// ---------------------------------------------------------------------------
// MinkUNet (dense masked emulation) on gfx1250 — round 4.
// Implicit-GEMM conv on V_WMMA_F32_16X16X32_F16, MODE-templated:
//   * 4 waves/block, each wave: one 16-voxel M-tile x TWO 16-cout N-tiles
//   * double-buffered LDS weight stage -> ONE s_barrier per K-chunk
//   * counter-based tap x cin-chunk loop (no runtime div/mod, no mode branch)
//   * 32-bit offset arithmetic (no 64-bit MULs in the hot loop)
//   * A: 4x guarded float4 loads (global_load_b128) + packed f32->f16 cvt
//   * weights: 2x coalesced float4 loads, f16 pairs packed along k,
//     4x b32 ds-stores into transposed [n][k] image; reads ds_load_b128
//   * global_prefetch_b8 on the weight stream (2 chunks ahead)
//   * fused epilogue: (+bias) -> BN -> (+identity) -> ReLU -> *mask
// ---------------------------------------------------------------------------

typedef __attribute__((ext_vector_type(16))) _Float16 v16h;
typedef __attribute__((ext_vector_type(8)))  _Float16 v8h;
typedef __attribute__((ext_vector_type(8)))  float    v8f;

#define NW 4            // waves per block (4 M-tiles share one weight stage)
#define LWS 40          // LDS row stride in halves (16B-aligned rows, padded)

struct ConvDesc {
  const float* __restrict__ x;     // input activations, dense [Din^3][Cin]
  const float* __restrict__ w;     // weights, row-major [taps*Cin][Cout]
  const float* __restrict__ bias;  // optional pre-BN bias (shortcut db)
  const float* __restrict__ g;
  const float* __restrict__ bb;
  const float* __restrict__ m;
  const float* __restrict__ v;     // BN running stats
  const float* __restrict__ mask;  // output-res occupancy mask, or null
  const float* __restrict__ idn;   // residual added pre-ReLU, or null
  float* __restrict__ out;         // out[voxel*cstride + coff + n]
  int sDin, sDout;                 // log2 cube dims
  int Cin, Cout;
  int relu;
  int coff, cstride, idn_cs;
};

// MODE: 0 = 3x3x3 SAME, 1 = 2x2x2 stride2 VALID, 2 = 1x1x1, 3 = deconv (tap = blockIdx.z)
template <int MODE>
__global__ __launch_bounds__(128)
void conv_wmma(ConvDesc d) {
  __shared__ __align__(16) _Float16 lws[2][32 * LWS];

  const int tid  = threadIdx.x;
  const int lane = tid & 31;
  const int wave = tid >> 5;
  const int half = lane >> 4;       // K-half selector (ISA 16-bit layouts)
  const int l16  = lane & 15;       // M row (A) / N col (B,C,D)
  const unsigned nbase = blockIdx.y << 5;        // 32-wide N super-tile
  const bool t1 = (int)(nbase + 16) < d.Cout;    // block-uniform 2nd N-tile

  const int sDm   = (MODE == 3) ? d.sDin : d.sDout;
  const int Mmask = (1 << sDm) - 1;
  const int Din   = 1 << d.sDin;

  const int mtile = blockIdx.x * NW + wave;
  const int mlin  = (mtile << 4) + l16;
  const int mw = mlin & Mmask;
  const int mh = (mlin >> sDm) & Mmask;
  const int md = mlin >> (2 * sDm);

  const int taps     = (MODE == 0) ? 27 : (MODE == 1) ? 8 : 1;
  const int wtapbase = (MODE == 3) ? (int)blockIdx.z * d.Cin : 0;
  const int total    = taps * ((d.Cin + 31) >> 5);

  // weight staging ownership: thread -> 2 k-rows, 4 n-cols
  const int wk2 = (tid & 15) << 1;                     // even k: 0,2,..,30
  const unsigned wng = (unsigned)(tid >> 4) << 2;      // n group: 0,4,..,28
  const bool wnok = (int)(nbase + wng) < d.Cout;       // uniform per 4-group

  auto stage = [&](int tap, int c0, int buf) {
    const int kk = c0 + wk2;
    float4 wa = make_float4(0.f, 0.f, 0.f, 0.f), wb = wa;
    const unsigned row = (unsigned)(wtapbase + tap * d.Cin + kk) * (unsigned)d.Cout
                       + nbase + wng;
    if (wnok & (kk < d.Cin))     wa = *(const float4*)(d.w + row);
    if (wnok & (kk + 1 < d.Cin)) wb = *(const float4*)(d.w + row + (unsigned)d.Cout);
    const float* pa = (const float*)&wa;
    const float* pb = (const float*)&wb;
#pragma unroll
    for (int j = 0; j < 4; ++j) {
      const unsigned short lo = __builtin_bit_cast(unsigned short, (_Float16)pa[j]);
      const unsigned short hi = __builtin_bit_cast(unsigned short, (_Float16)pb[j]);
      *(unsigned int*)&lws[buf][(wng + j) * LWS + wk2] = ((unsigned)hi << 16) | lo;
    }
  };
  auto advW = [&](int& tap, int& c0) { c0 += 32; if (c0 >= d.Cin) { c0 = 0; ++tap; } };

  // pipeline prologue: stage chunk 0 into buffer 0
  int stap = 0, sc0 = 0;
  stage(0, 0, 0);
  advW(stap, sc0);
  __syncthreads();

  int ctap = 0, cc0 = 0;            // compute-side counters
  int cdx = 0, cdy = 0, cdz = 0;    // MODE-0 tap offsets (incremental)
  v8f acc0 = {}, acc1 = {};

  for (int it = 0; it < total; ++it) {
    const int buf = it & 1;
    if (it + 1 < total) {
      stage(stap, sc0, buf ^ 1);
      advW(stap, sc0);
      if (it + 2 < total)   // keep the weight stream warm in GL2
        __builtin_prefetch(d.w + (unsigned)(wtapbase + stap * d.Cin + sc0) * (unsigned)d.Cout
                               + nbase, 0, 1);
    }

    // ---- per-lane input voxel base for current tap ----------------------
    int di, hi, wi; bool ok = true;
    if (MODE == 0) {
      di = md + cdz - 1; hi = mh + cdy - 1; wi = mw + cdx - 1;
      ok = ((unsigned)di < (unsigned)Din) & ((unsigned)hi < (unsigned)Din)
         & ((unsigned)wi < (unsigned)Din);
    } else if (MODE == 1) {
      di = 2 * md + (ctap >> 2); hi = 2 * mh + ((ctap >> 1) & 1); wi = 2 * mw + (ctap & 1);
    } else { di = md; hi = mh; wi = mw; }
    const unsigned vox = (unsigned)((((di << d.sDin) + hi) << d.sDin) + wi);
    const float* xrow = d.x + (ok ? vox * (unsigned)d.Cin : 0u);

    // ---- A fragment: two contiguous 8-float runs, float4 loads ----------
    const int r0 = cc0 + (half << 3);
    const int r1 = r0 + 16;
    float4 A0 = make_float4(0.f, 0.f, 0.f, 0.f), A1 = A0, B0 = A0, B1 = A0;
    if (ok & (r0 < d.Cin)) { A0 = *(const float4*)(xrow + r0); A1 = *(const float4*)(xrow + r0 + 4); }
    if (ok & (r1 < d.Cin)) { B0 = *(const float4*)(xrow + r1); B1 = *(const float4*)(xrow + r1 + 4); }
    v16h a;
    a[0]  = (_Float16)A0.x; a[1]  = (_Float16)A0.y; a[2]  = (_Float16)A0.z; a[3]  = (_Float16)A0.w;
    a[4]  = (_Float16)A1.x; a[5]  = (_Float16)A1.y; a[6]  = (_Float16)A1.z; a[7]  = (_Float16)A1.w;
    a[8]  = (_Float16)B0.x; a[9]  = (_Float16)B0.y; a[10] = (_Float16)B0.z; a[11] = (_Float16)B0.w;
    a[12] = (_Float16)B1.x; a[13] = (_Float16)B1.y; a[14] = (_Float16)B1.z; a[15] = (_Float16)B1.w;

    // ---- B fragments from LDS (16B-aligned -> ds_load_b128) -------------
    {
      const _Float16* lp = &lws[buf][l16 * LWS + (half << 3)];
      v8h blo = *(const v8h*)lp;
      v8h bhi = *(const v8h*)(lp + 16);
      v16h b;
#pragma unroll
      for (int j = 0; j < 8; ++j) { b[j] = blo[j]; b[8 + j] = bhi[j]; }
      acc0 = __builtin_amdgcn_wmma_f32_16x16x32_f16(false, a, false, b,
                                                    (short)0, acc0, false, false);
    }
    if (t1) {  // block-uniform: EXEC stays all-ones around WMMA
      const _Float16* lp = &lws[buf][(l16 + 16) * LWS + (half << 3)];
      v8h blo = *(const v8h*)lp;
      v8h bhi = *(const v8h*)(lp + 16);
      v16h b;
#pragma unroll
      for (int j = 0; j < 8; ++j) { b[j] = blo[j]; b[8 + j] = bhi[j]; }
      acc1 = __builtin_amdgcn_wmma_f32_16x16x32_f16(false, a, false, b,
                                                    (short)0, acc1, false, false);
    }

    __syncthreads();   // one barrier/chunk: stage(it+1) done, reads of buf done
    cc0 += 32;
    if (cc0 >= d.Cin) {
      cc0 = 0; ++ctap;
      if (MODE == 0) {
        ++cdx; if (cdx == 3) { cdx = 0; ++cdy; if (cdy == 3) { cdy = 0; ++cdz; } }
      }
    }
  }

  // ---- fused epilogue (per N-tile): bias -> BN -> +idn -> ReLU -> *mask --
  const int lin0 = mtile << 4;
  const int tz = (MODE == 3) ? ((blockIdx.z >> 2) & 1) : 0;
  const int ty = (MODE == 3) ? ((blockIdx.z >> 1) & 1) : 0;
  const int tx = (MODE == 3) ? (blockIdx.z & 1) : 0;
#pragma unroll
  for (int t = 0; t < 2; ++t) {
    if (t == 1 && !t1) break;
    const unsigned n = nbase + (t << 4) + l16;
    const v8f acc = t ? acc1 : acc0;
    const float scale = d.g[n] * rsqrtf(d.v[n] + 1e-5f);
    const float shift = d.bb[n] - d.m[n] * scale;
    const float bia   = d.bias ? d.bias[n] : 0.f;
#pragma unroll
    for (int r = 0; r < 8; ++r) {
      const int mm  = r + (half << 3);        // C/D layout: VGPR r, half -> M
      const int lin = lin0 + mm;
      const int ow = lin & Mmask;
      const int oh = (lin >> sDm) & Mmask;
      const int od = lin >> (2 * sDm);
      unsigned outlin;
      if (MODE == 3) {
        outlin = (unsigned)(((((2 * od + tz) << d.sDout) + (2 * oh + ty)) << d.sDout)
                            + 2 * ow + tx);
      } else {
        outlin = (unsigned)((((od << sDm) + oh) << sDm) + ow);
      }
      float val = (acc[r] + bia) * scale + shift;
      if (d.idn)  val += d.idn[outlin * (unsigned)d.idn_cs + n];
      if (d.relu) val = fmaxf(val, 0.f);
      if (d.mask) val *= d.mask[outlin];
      d.out[outlin * (unsigned)d.cstride + (unsigned)d.coff + n] = val;
    }
  }
}

// occupancy of stride-2 ks-2 window: 2x2x2 max-pool of the mask
__global__ void maskpool_k(const float* __restrict__ in, float* __restrict__ out, int sDo) {
  const int i = blockIdx.x * blockDim.x + threadIdx.x;
  const int tot = 1 << (3 * sDo);
  if (i >= tot) return;
  const int Mk = (1 << sDo) - 1;
  const int w = i & Mk, h = (i >> sDo) & Mk, dd = i >> (2 * sDo);
  const int sDi = sDo + 1;
  float mx = 0.f;
#pragma unroll
  for (int z = 0; z < 2; ++z)
#pragma unroll
    for (int y = 0; y < 2; ++y)
#pragma unroll
      for (int x = 0; x < 2; ++x)
        mx = fmaxf(mx, in[(unsigned)((((2 * dd + z) << sDi) + 2 * h + y) << sDi) + 2 * w + x]);
  out[i] = mx;
}

// strided channel copy (skip tensor -> concat buffer at channel offset)
__global__ void copych_k(const float* __restrict__ src, int scs,
                         float* __restrict__ dst, int dcs, int dcoff,
                         int sC, int nelem) {
  const int i = blockIdx.x * blockDim.x + threadIdx.x;
  if (i >= nelem) return;
  const unsigned v = (unsigned)i >> sC, c = (unsigned)i & ((1u << sC) - 1u);
  dst[v * (unsigned)dcs + (unsigned)dcoff + c] = src[v * (unsigned)scs + c];
}

// ---------------------------------------------------------------------------

struct BNP { const float *b, *g, *m, *v; };
struct ResP {
  BNP bn1, bn2, dbn;
  const float *db, *dw, *w1, *w2;
  bool has_sc;
};
struct StageP { BNP bn; ResP r0, r1; const float* w; };

static inline int ilog2(int x) { int s = 0; while ((1 << s) < x) ++s; return s; }

extern "C" void kernel_launch(void* const* d_in, const int* in_sizes, int n_in,
                              void* d_out, int out_size, void* d_ws, size_t ws_size,
                              hipStream_t stream) {
  (void)in_sizes; (void)n_in; (void)out_size; (void)ws_size;

  // ---- pytree leaf order (dicts flatten sorted by key) -------------------
  int cur = 2;
  auto F   = [&]() -> const float* { return (const float*)d_in[cur++]; };
  auto RBN = [&](BNP& bn) { bn.b = F(); bn.g = F(); bn.m = F(); bn.v = F(); };
  auto RRES = [&](ResP& r, bool sc) {
    RBN(r.bn1); RBN(r.bn2);
    r.has_sc = sc; r.db = nullptr; r.dw = nullptr;
    if (sc) { r.db = F(); RBN(r.dbn); r.dw = F(); }
    r.w1 = F(); r.w2 = F();
  };

  const float* X  = (const float*)d_in[0];
  const float* M0 = (const float*)d_in[1];

  BNP ci0bn, ci1bn;
  RBN(ci0bn); const float* ci0w = F();
  RBN(ci1bn); const float* ci1w = F();

  StageP st[4], up[4];
  const bool st_sc[4] = {true, true, true, false};
  for (int i = 0; i < 4; ++i) { RBN(st[i].bn); RRES(st[i].r0, st_sc[i]); RRES(st[i].r1, false); st[i].w = F(); }
  for (int i = 0; i < 4; ++i) { RBN(up[i].bn); RRES(up[i].r0, true);     RRES(up[i].r1, false); up[i].w = F(); }

  // ---- workspace carve-up (floats); ~40.5M floats (~162 MB) --------------
  float* ws = (float*)d_ws;
  size_t off = 0;
  auto alloc = [&](size_t nf) { float* p = ws + off; off += nf; return p; };
  float* m1 = alloc(32768);      // 32^3
  float* m2 = alloc(4096);       // 16^3
  float* m3 = alloc(512);        // 8^3
  float* m4 = alloc(64);         // 4^3
  float* x0 = alloc(8388608);    // 64^3*32
  float* x1 = alloc(2097152);    // 32^3*64
  float* x2 = alloc(524288);     // 16^3*128
  float* x3 = alloc(131072);     // 8^3*256
  float* x4 = alloc(16384);      // 4^3*256
  float* idn = alloc(4194304);   // max shortcut: 64^3*16
  float* A  = alloc(12582912);   // 64^3*48 ping
  float* B  = alloc(12582912);   // 64^3*48 pong

  // ---- launch helpers ----------------------------------------------------
  auto conv = [&](int mode, const float* x, const float* w, const float* bias,
                  BNP bn, const float* mask, const float* id, int id_cs,
                  float* out, int Din, int Dout, int Cin, int Cout,
                  int relu, int coff, int cs) {
    ConvDesc dsc;
    dsc.x = x; dsc.w = w; dsc.bias = bias;
    dsc.g = bn.g; dsc.bb = bn.b; dsc.m = bn.m; dsc.v = bn.v;
    dsc.mask = mask; dsc.idn = id; dsc.out = out;
    dsc.sDin = ilog2(Din); dsc.sDout = ilog2(Dout);
    dsc.Cin = Cin; dsc.Cout = Cout;
    dsc.relu = relu; dsc.coff = coff; dsc.cstride = cs;
    dsc.idn_cs = id_cs;
    const int Mvox = (mode == 3) ? Din * Din * Din : Dout * Dout * Dout;
    dim3 grid(Mvox / (16 * NW), (Cout + 31) / 32, (mode == 3) ? 8 : 1);
    switch (mode) {
      case 0: conv_wmma<0><<<grid, 32 * NW, 0, stream>>>(dsc); break;
      case 1: conv_wmma<1><<<grid, 32 * NW, 0, stream>>>(dsc); break;
      case 2: conv_wmma<2><<<grid, 32 * NW, 0, stream>>>(dsc); break;
      default: conv_wmma<3><<<grid, 32 * NW, 0, stream>>>(dsc); break;
    }
  };

  // resblock: h = relu(bn1(conv3(in)))*m ; h = bn2(conv3(h)) ; out = relu(h+idn)*m
  auto res = [&](ResP& p, int Cin, int Cout, int D, const float* mask,
                 float* in, float* tmp, float* out) {
    const float* idp; int idcs;
    if (p.has_sc) {
      conv(2, in, p.dw, p.db, p.dbn, nullptr, nullptr, 0, idn, D, D, Cin, Cout, 0, 0, Cout);
      idp = idn; idcs = Cout;
    } else { idp = in; idcs = Cin; }
    conv(0, in,  p.w1, nullptr, p.bn1, mask, nullptr, 0,   tmp, D, D, Cin,  Cout, 1, 0, Cout);
    conv(0, tmp, p.w2, nullptr, p.bn2, mask, idp,    idcs, out, D, D, Cout, Cout, 1, 0, Cout);
  };

  // ---- mask pyramid ------------------------------------------------------
  maskpool_k<<<(32768 + 255) / 256, 256, 0, stream>>>(M0, m1, 5);
  maskpool_k<<<(4096  + 255) / 256, 256, 0, stream>>>(m1, m2, 4);
  maskpool_k<<<(512   + 255) / 256, 256, 0, stream>>>(m2, m3, 3);
  maskpool_k<<<(64    +  63) /  64,  64, 0, stream>>>(m3, m4, 2);

  // ---- stem --------------------------------------------------------------
  conv(0, X, ci0w, nullptr, ci0bn, M0, nullptr, 0, A,  64, 64, 16, 32, 1, 0, 32);
  conv(0, A, ci1w, nullptr, ci1bn, M0, nullptr, 0, x0, 64, 64, 32, 32, 1, 0, 32);

  // ---- encoder -----------------------------------------------------------
  conv(1, x0, st[0].w, nullptr, st[0].bn, m1, nullptr, 0, A, 64, 32, 32, 32, 1, 0, 32);
  res(st[0].r0, 32,  64, 32, m1, A, B, A);
  res(st[0].r1, 64,  64, 32, m1, A, B, x1);

  conv(1, x1, st[1].w, nullptr, st[1].bn, m2, nullptr, 0, A, 32, 16, 64, 64, 1, 0, 64);
  res(st[1].r0, 64,  128, 16, m2, A, B, A);
  res(st[1].r1, 128, 128, 16, m2, A, B, x2);

  conv(1, x2, st[2].w, nullptr, st[2].bn, m3, nullptr, 0, A, 16, 8, 128, 128, 1, 0, 128);
  res(st[2].r0, 128, 256, 8, m3, A, B, A);
  res(st[2].r1, 256, 256, 8, m3, A, B, x3);

  conv(1, x3, st[3].w, nullptr, st[3].bn, m4, nullptr, 0, A, 8, 4, 256, 256, 1, 0, 256);
  res(st[3].r0, 256, 256, 4, m4, A, B, A);
  res(st[3].r1, 256, 256, 4, m4, A, B, x4);

  // ---- decoder -----------------------------------------------------------
  conv(3, x4, up[0].w, nullptr, up[0].bn, m3, nullptr, 0, A, 4, 8, 256, 128, 1, 0, 384);
  copych_k<<<(512 * 256 + 255) / 256, 256, 0, stream>>>(x3, 256, A, 384, 128, 8, 512 * 256);
  res(up[0].r0, 384, 128, 8, m3, A, B, A);
  res(up[0].r1, 128, 128, 8, m3, A, B, A);

  conv(3, A, up[1].w, nullptr, up[1].bn, m2, nullptr, 0, B, 8, 16, 128, 64, 1, 0, 192);
  copych_k<<<(4096 * 128 + 255) / 256, 256, 0, stream>>>(x2, 128, B, 192, 64, 7, 4096 * 128);
  res(up[1].r0, 192, 64, 16, m2, B, A, B);
  res(up[1].r1, 64,  64, 16, m2, B, A, B);

  conv(3, B, up[2].w, nullptr, up[2].bn, m1, nullptr, 0, A, 16, 32, 64, 32, 1, 0, 96);
  copych_k<<<(32768 * 64 + 255) / 256, 256, 0, stream>>>(x1, 64, A, 96, 32, 6, 32768 * 64);
  res(up[2].r0, 96, 32, 32, m1, A, B, A);
  res(up[2].r1, 32, 32, 32, m1, A, B, A);

  conv(3, A, up[3].w, nullptr, up[3].bn, M0, nullptr, 0, B, 32, 64, 32, 16, 1, 0, 48);
  copych_k<<<(262144 * 32 + 255) / 256, 256, 0, stream>>>(x0, 32, B, 48, 16, 5, 262144 * 32);
  res(up[3].r0, 48, 16, 64, M0, B, A, B);
  res(up[3].r1, 16, 16, 64, M0, B, A, (float*)d_out);
}
